// MultiHeadDotProductAttention_68899865362780
// MI455X (gfx1250) — compile-verified, matrix operand order checked by reading, not run on hardware
//
#include <hip/hip_runtime.h>
#include <hip/hip_bf16.h>

// MI455X (gfx1250) fused causal MHA with QK-LayerNorm.
// B=2, T=4096, E=512, H=8, D=64. Matmuls via v_wmma_f32_16x16x32_f16;
// K/V tiles staged with global_load_async_to_lds_b128 (ASYNCcnt) when available.

#define BB 2
#define TT 4096
#define EE 512
#define HH 8
#define DD 64
#define HD 512

#if defined(__AMDGCN__) && __has_builtin(__builtin_amdgcn_global_load_async_to_lds_b128)
#define USE_ASYNC_LDS 1
#define AS1 __attribute__((address_space(1)))
#define AS3 __attribute__((address_space(3)))
typedef int v4i __attribute__((vector_size(16)));   // matches builtin param type
#endif
#ifndef USE_ASYNC_LDS
#define USE_ASYNC_LDS 0
#endif

typedef __attribute__((ext_vector_type(16))) _Float16 v16h;
typedef __attribute__((ext_vector_type(8)))  _Float16 v8h;
typedef __attribute__((ext_vector_type(8)))  float    v8f;

// 16-bit A/B fragment loader. Per ISA 7.12.2: lane L holds row/col L%16;
// lanes 0-15 hold K = {0..7, 16..23}, lanes 16-31 hold K = {8..15, 24..31}.
// kb = (lane&16) ? 8 : 0 -> two contiguous 16-byte runs at p+kb and p+kb+16.
__device__ inline v16h frag16(const _Float16* p, int kb) {
  union { v16h v; v8h h[2]; } u;
  u.h[0] = *(const v8h*)(p + kb);
  u.h[1] = *(const v8h*)(p + kb + 16);
  return u.v;
}

__device__ inline v8f wmma(v16h a, v16h b, v8f c) {
  return __builtin_amdgcn_wmma_f32_16x16x32_f16(
      /*neg_a=*/false, a, /*neg_b=*/false, b,
      /*c_mod=*/(short)0, c, /*reuse_a=*/false, /*reuse_b=*/false);
}

// ---------------- Stage 0: weight transpose + f32->f16 ----------------
// out[c][r] = (f16) in[r][c]  (so B-fragments read contiguous K per lane)
__global__ void k_transpose_f16(const float* __restrict__ in,
                                _Float16* __restrict__ out, int rows, int cols) {
  int idx = blockIdx.x * blockDim.x + threadIdx.x;
  if (idx < rows * cols) {
    int r = idx / cols, c = idx - r * cols;
    out[(size_t)c * rows + r] = (_Float16)in[idx];
  }
}

// ---------------- Stage 1: QKV projection + QK LayerNorm ----------------
// One block = 16 token rows; 8 waves, wave w owns output cols [w*64, w*64+64).
__global__ __launch_bounds__(256) void k_qkv_ln(
    const float* __restrict__ Xq, const float* __restrict__ Xkv,
    const _Float16* __restrict__ WqT, const _Float16* __restrict__ WkT,
    const _Float16* __restrict__ WvT,
    const float* __restrict__ qw, const float* __restrict__ qb,
    const float* __restrict__ kw, const float* __restrict__ kbias,
    _Float16* __restrict__ qh, _Float16* __restrict__ kh,
    _Float16* __restrict__ vT) {
  __shared__ _Float16 xq[16][HD];   // reused later for raw-q staging before LN
  __shared__ _Float16 xk[16][HD];
  __shared__ float stats[4][16];

  const int row0 = blockIdx.x * 16;          // global row in [0, B*T)
  const int b = row0 / TT;
  const int t0 = row0 - b * TT;
  const int wave = threadIdx.x >> 5;
  const int lane = threadIdx.x & 31;
  const int half8 = (lane & 16) ? 8 : 0;

  // stage 16 rows of each input to LDS as f16 (coalesced float4 loads)
  for (int i = threadIdx.x; i < 16 * EE / 4; i += blockDim.x) {
    int r = i >> 7;
    int c = (i & 127) * 4;
    float4 a = *(const float4*)(Xq  + (size_t)(row0 + r) * EE + c);
    float4 d = *(const float4*)(Xkv + (size_t)(row0 + r) * EE + c);
    xq[r][c] = (_Float16)a.x; xq[r][c+1] = (_Float16)a.y;
    xq[r][c+2] = (_Float16)a.z; xq[r][c+3] = (_Float16)a.w;
    xk[r][c] = (_Float16)d.x; xk[r][c+1] = (_Float16)d.y;
    xk[r][c+2] = (_Float16)d.z; xk[r][c+3] = (_Float16)d.w;
  }
  __syncthreads();

  const int c0 = wave * 64;
  v8f aq[4] = {}, ak[4] = {}, av[4] = {};
  for (int k0 = 0; k0 < EE; k0 += 32) {
    v16h aqf = frag16(&xq[lane & 15][k0], half8);
    v16h akf = frag16(&xk[lane & 15][k0], half8);
#pragma unroll
    for (int t = 0; t < 4; ++t) {
      int n = c0 + t * 16 + (lane & 15);
      aq[t] = wmma(aqf, frag16(WqT + (size_t)n * EE + k0, half8), aq[t]);
      ak[t] = wmma(akf, frag16(WkT + (size_t)n * EE + k0, half8), ak[t]);
      av[t] = wmma(akf, frag16(WvT + (size_t)n * EE + k0, half8), av[t]);
    }
  }

  // V needs no LN: store directly, transposed to [B,H,D,T].
  // C-layout: lane holds 8 consecutive token rows -> one 16B store per tile.
#pragma unroll
  for (int t = 0; t < 4; ++t) {
    int col = c0 + t * 16 + (lane & 15);
    int hh = col >> 6, d = col & 63;
    v8h pk;
#pragma unroll
    for (int r = 0; r < 8; ++r) pk[r] = (_Float16)av[t][r];
    *(v8h*)(vT + (((size_t)(b * HH + hh)) * DD + d) * TT + t0 + half8) = pk;
  }

  __syncthreads();   // xq/xk fully consumed; reuse for raw q/k tiles
#pragma unroll
  for (int t = 0; t < 4; ++t) {
    int col = c0 + t * 16 + (lane & 15);
#pragma unroll
    for (int r = 0; r < 8; ++r) {
      xq[r + half8][col] = (_Float16)aq[t][r];
      xk[r + half8][col] = (_Float16)ak[t][r];
    }
  }
  __syncthreads();

  // per-row stats over 512 dims; 16 threads per row, shuffle-reduce
  {
    int rr = threadIdx.x >> 4;
    int j  = threadIdx.x & 15;
    float sq = 0, sqq = 0, sk = 0, skk = 0;
    for (int i = j; i < HD; i += 16) {
      float x = (float)xq[rr][i]; sq += x; sqq += x * x;
      float y = (float)xk[rr][i]; sk += y; skk += y * y;
    }
#pragma unroll
    for (int off = 1; off < 16; off <<= 1) {
      sq  += __shfl_xor(sq,  off, 32);
      sqq += __shfl_xor(sqq, off, 32);
      sk  += __shfl_xor(sk,  off, 32);
      skk += __shfl_xor(skk, off, 32);
    }
    if (j == 0) {
      float mq = sq * (1.0f / HD);
      float mk = sk * (1.0f / HD);
      stats[0][rr] = mq;
      stats[1][rr] = rsqrtf(sqq * (1.0f / HD) - mq * mq + 1e-5f);
      stats[2][rr] = mk;
      stats[3][rr] = rsqrtf(skk * (1.0f / HD) - mk * mk + 1e-5f);
    }
  }
  __syncthreads();

  // normalize; fold attention scale 1/sqrt(D)=0.125 into q; write [B,H,T,D] f16
  for (int i = threadIdx.x; i < 16 * 64; i += blockDim.x) {
    int r  = i >> 6;
    int cc = (i & 63) * 8;
    float mq = stats[0][r], rq = stats[1][r];
    float mk = stats[2][r], rk = stats[3][r];
    v8h oq, ok;
#pragma unroll
    for (int u = 0; u < 8; ++u) {
      int col = cc + u;
      oq[u] = (_Float16)((((float)xq[r][col] - mq) * rq * qw[col] + qb[col]) * 0.125f);
      ok[u] = (_Float16)((((float)xk[r][col] - mk) * rk * kw[col] + kbias[col]));
    }
    int hh = cc >> 6, d = cc & 63;
    size_t o = (((size_t)(b * HH + hh)) * TT + (t0 + r)) * DD + d;
    *(v8h*)(qh + o) = oq;
    *(v8h*)(kh + o) = ok;
  }
}

// ---------------- Stage 2: causal flash attention ----------------
// Block = 64 q rows of one (b,h); 4 waves x 16 rows. Online softmax.
// K/V tiles double-buffered in LDS, filled by async-to-LDS DMA (ASYNCcnt).
__global__ __launch_bounds__(128) void k_attn(
    const _Float16* __restrict__ qh, const _Float16* __restrict__ kh,
    const _Float16* __restrict__ vT, _Float16* __restrict__ attn) {
  __shared__ _Float16 Ks[2][64][64];    // [buf][p_local][d]
  __shared__ _Float16 Vs[2][64][64];    // [buf][d][p_local]
  __shared__ _Float16 Ps[4][16][64];    // per-wave P / O staging

  const int qt = blockIdx.x & 63;
  const int h  = (blockIdx.x >> 6) & 7;
  const int b  = blockIdx.x >> 9;
  const int wave = threadIdx.x >> 5;
  const int lane = threadIdx.x & 31;
  const int half8 = (lane & 16) ? 8 : 0;
  const int q0 = qt * 64 + wave * 16;

  const _Float16* qrow = qh + (((size_t)(b * HH + h)) * TT + q0 + (lane & 15)) * DD;
  v16h qf0 = frag16(qrow, half8);        // d 0..31 (scale pre-folded)
  v16h qf1 = frag16(qrow + 32, half8);   // d 32..63

  v8f o[4] = {};
  v8f m, l;
#pragma unroll
  for (int r = 0; r < 8; ++r) { m[r] = -1e30f; l[r] = 0.0f; }

  const _Float16* kbase = kh + ((size_t)(b * HH + h)) * TT * DD;
  const _Float16* vbase = vT + ((size_t)(b * HH + h)) * DD * TT;
  const int nkb = qt + 1;                // causal: only blocks <= diagonal

  // stage a 64x64 K tile and 64x64 V^T tile into LDS buffer `buf`
  auto stage = [&](int p0, int buf) {
    for (int i = threadIdx.x; i < 64 * 64 / 8; i += blockDim.x) {
      int row = i >> 3;
      int c = (i & 7) * 8;
#if USE_ASYNC_LDS
      __builtin_amdgcn_global_load_async_to_lds_b128(
          (AS1 v4i*)(kbase + (size_t)(p0 + row) * DD + c),
          (AS3 v4i*)&Ks[buf][row][c], 0, 0);
      __builtin_amdgcn_global_load_async_to_lds_b128(
          (AS1 v4i*)(vbase + (size_t)row * TT + p0 + c),
          (AS3 v4i*)&Vs[buf][row][c], 0, 0);
#else
      *(v8h*)&Ks[buf][row][c] = *(const v8h*)(kbase + (size_t)(p0 + row) * DD + c);
      *(v8h*)&Vs[buf][row][c] = *(const v8h*)(vbase + (size_t)row * TT + p0 + c);
#endif
    }
  };

  stage(0, 0);
  for (int ib = 0; ib < nkb; ++ib) {
    const int p0 = ib * 64;
    const int cur = ib & 1;
#if USE_ASYNC_LDS
    asm volatile("s_wait_asynccnt 0x0" ::: "memory");  // our tile landed in LDS
#endif
    __syncthreads();  // publishes buf `cur`; orders prev reads of buf cur^1
    if (ib + 1 < nkb) stage(p0 + 64, cur ^ 1);  // overlap next DMA with compute

    // S = Q * K^T  (16x64, 8 WMMAs)
    v8f s[4] = {};
#pragma unroll
    for (int t = 0; t < 4; ++t) {
      const _Float16* krow = &Ks[cur][t * 16 + (lane & 15)][0];
      s[t] = wmma(qf0, frag16(krow, half8), s[t]);
      s[t] = wmma(qf1, frag16(krow + 32, half8), s[t]);
    }
    // causal mask (only near the diagonal); finite sentinel avoids inf-inf
    if (p0 + 63 > q0) {
      int n = lane & 15;
#pragma unroll
      for (int t = 0; t < 4; ++t) {
        int pk = p0 + t * 16 + n;
#pragma unroll
        for (int r = 0; r < 8; ++r)
          if (pk > q0 + r + half8) s[t][r] = -1e30f;
      }
    }
    // online softmax: rows live in 16-lane halves -> xor-shuffle reduce
    v8f rmax = s[0];
#pragma unroll
    for (int t = 1; t < 4; ++t)
#pragma unroll
      for (int r = 0; r < 8; ++r) rmax[r] = fmaxf(rmax[r], s[t][r]);
#pragma unroll
    for (int off = 1; off < 16; off <<= 1)
#pragma unroll
      for (int r = 0; r < 8; ++r) rmax[r] = fmaxf(rmax[r], __shfl_xor(rmax[r], off, 32));
    v8f alpha;
#pragma unroll
    for (int r = 0; r < 8; ++r) {
      float mn = fmaxf(m[r], rmax[r]);
      alpha[r] = __expf(m[r] - mn);
      m[r] = mn;
    }
    v8f rsum = {};
#pragma unroll
    for (int t = 0; t < 4; ++t)
#pragma unroll
      for (int r = 0; r < 8; ++r) { s[t][r] = __expf(s[t][r] - m[r]); rsum[r] += s[t][r]; }
#pragma unroll
    for (int off = 1; off < 16; off <<= 1)
#pragma unroll
      for (int r = 0; r < 8; ++r) rsum[r] += __shfl_xor(rsum[r], off, 32);
#pragma unroll
    for (int r = 0; r < 8; ++r) l[r] = l[r] * alpha[r] + rsum[r];
#pragma unroll
    for (int t = 0; t < 4; ++t)
#pragma unroll
      for (int r = 0; r < 8; ++r) o[t][r] *= alpha[r];

    // C-layout -> A-layout for P via per-wave LDS, then O += P * V
#pragma unroll
    for (int t = 0; t < 4; ++t)
#pragma unroll
      for (int r = 0; r < 8; ++r)
        Ps[wave][r + half8][t * 16 + (lane & 15)] = (_Float16)s[t][r];
    const _Float16* prow = &Ps[wave][lane & 15][0];
    v16h pa0 = frag16(prow, half8);
    v16h pa1 = frag16(prow + 32, half8);
#pragma unroll
    for (int t = 0; t < 4; ++t) {
      const _Float16* vrow = &Vs[cur][t * 16 + (lane & 15)][0];
      o[t] = wmma(pa0, frag16(vrow, half8), o[t]);
      o[t] = wmma(pa1, frag16(vrow + 32, half8), o[t]);
    }
  }

  // finalize O = O / l, stage to LDS, coalesced f16 write to [B,T,H*D]
  v8f inv;
#pragma unroll
  for (int r = 0; r < 8; ++r) inv[r] = 1.0f / l[r];
#pragma unroll
  for (int t = 0; t < 4; ++t)
#pragma unroll
    for (int r = 0; r < 8; ++r)
      Ps[wave][r + half8][t * 16 + (lane & 15)] = (_Float16)(o[t][r] * inv[r]);
  {
    int row = lane >> 1;
    int cc = (lane & 1) * 32;
    size_t base = ((size_t)b * TT + q0 + row) * HD + (size_t)h * 64 + cc;
#pragma unroll
    for (int j = 0; j < 4; ++j)
      *(v8h*)(attn + base + j * 8) = *(const v8h*)&Ps[wave][row][cc + j * 8];
  }
}

// ---------------- Stage 3: output projection ----------------
__global__ __launch_bounds__(256) void k_outproj(
    const _Float16* __restrict__ attn, const _Float16* __restrict__ WoT,
    float* __restrict__ out) {
  __shared__ _Float16 xa[16][HD];
  __shared__ float ob[16][EE];
  const int row0 = blockIdx.x * 16;
  const int wave = threadIdx.x >> 5;
  const int lane = threadIdx.x & 31;
  const int half8 = (lane & 16) ? 8 : 0;

  // pure f16 copy -> async DMA straight to LDS (no VGPR round-trip)
  for (int i = threadIdx.x; i < 16 * HD / 8; i += blockDim.x) {
    int r = i >> 6;
    int c = (i & 63) * 8;
#if USE_ASYNC_LDS
    __builtin_amdgcn_global_load_async_to_lds_b128(
        (AS1 v4i*)(attn + (size_t)(row0 + r) * HD + c),
        (AS3 v4i*)&xa[r][c], 0, 0);
#else
    *(v8h*)&xa[r][c] = *(const v8h*)(attn + (size_t)(row0 + r) * HD + c);
#endif
  }
#if USE_ASYNC_LDS
  asm volatile("s_wait_asynccnt 0x0" ::: "memory");
#endif
  __syncthreads();

  const int c0 = wave * 64;
  v8f acc[4] = {};
  for (int k0 = 0; k0 < HD; k0 += 32) {
    v16h a = frag16(&xa[lane & 15][k0], half8);
#pragma unroll
    for (int t = 0; t < 4; ++t) {
      int n = c0 + t * 16 + (lane & 15);
      acc[t] = wmma(a, frag16(WoT + (size_t)n * HD + k0, half8), acc[t]);
    }
  }
#pragma unroll
  for (int t = 0; t < 4; ++t)
#pragma unroll
    for (int r = 0; r < 8; ++r)
      ob[r + half8][c0 + t * 16 + (lane & 15)] = acc[t][r];
  __syncthreads();
  for (int i = threadIdx.x; i < 16 * EE / 4; i += blockDim.x) {
    int r = i >> 7;
    int c = (i & 127) * 4;
    *(float4*)(out + (size_t)(row0 + r) * EE + c) = *(const float4*)&ob[r][c];
  }
}

// ---------------- host ----------------
extern "C" void kernel_launch(void* const* d_in, const int* in_sizes, int n_in,
                              void* d_out, int out_size, void* d_ws, size_t ws_size,
                              hipStream_t stream) {
  (void)in_sizes; (void)n_in; (void)out_size; (void)ws_size;
  const float* Xq  = (const float*)d_in[0];
  const float* Xkv = (const float*)d_in[1];
  // d_in[2] = mask (causal; computed analytically in-kernel)
  const float* Wq = (const float*)d_in[3];
  const float* Wk = (const float*)d_in[4];
  const float* Wv = (const float*)d_in[5];
  const float* Wo = (const float*)d_in[6];
  const float* qw = (const float*)d_in[7];
  const float* qb = (const float*)d_in[8];
  const float* kw = (const float*)d_in[9];
  const float* kb = (const float*)d_in[10];
  float* out = (float*)d_out;

  _Float16* p = (_Float16*)d_ws;                 // ~36 MB workspace
  _Float16* WqT = p; p += (size_t)EE * HD;
  _Float16* WkT = p; p += (size_t)EE * HD;
  _Float16* WvT = p; p += (size_t)EE * HD;
  _Float16* WoT = p; p += (size_t)HD * EE;
  _Float16* qh  = p; p += (size_t)BB * HH * TT * DD;
  _Float16* kh  = p; p += (size_t)BB * HH * TT * DD;
  _Float16* vT  = p; p += (size_t)BB * HH * DD * TT;
  _Float16* at  = p; p += (size_t)BB * TT * HD;

  const int tp = 256;
  const int nW = EE * HD;
  k_transpose_f16<<<(nW + tp - 1) / tp, tp, 0, stream>>>(Wq, WqT, EE, HD);
  k_transpose_f16<<<(nW + tp - 1) / tp, tp, 0, stream>>>(Wk, WkT, EE, HD);
  k_transpose_f16<<<(nW + tp - 1) / tp, tp, 0, stream>>>(Wv, WvT, EE, HD);
  k_transpose_f16<<<(nW + tp - 1) / tp, tp, 0, stream>>>(Wo, WoT, HD, EE);

  k_qkv_ln<<<BB * TT / 16, 256, 0, stream>>>(Xq, Xkv, WqT, WkT, WvT,
                                             qw, qb, kw, kb, qh, kh, vT);
  k_attn<<<BB * HH * (TT / 64), 128, 0, stream>>>(qh, kh, vT, at);
  k_outproj<<<BB * TT / 16, 256, 0, stream>>>(at, WoT, out);
}
